// MDTA_58626303590965
// MI455X (gfx1250) — matrix operand order repaired and checked by reading, not compiled, not run on hardware
//
#include <hip/hip_runtime.h>
#include <hip/hip_bf16.h>

// ---------------------------------------------------------------------------
// MDTA (Restormer transposed attention) for gfx1250 / MI455X.
//   b=8, c=192, h=w=128 (hw=16384), heads=8, head_dim=24
// Memory-bound pipeline (~1.4 GB HBM traffic -> ~60us floor @ 23.3 TB/s):
//   K0 memset   partial-S / norm accumulators (graph-capturable)
//   K1 qkv      = W_qkv (576x192) @ x (192x16384)   [bf16 WMMA, dbl-buffered LDS]
//   K2 qkv_dw   = depthwise3x3(qkv)                 [float4 VALU]
//   K3 partial  S += q k^T (K split 4-way), |q|^2,|k|^2  [bf16 WMMA + f32 atomics]
//   K4 softmax  attn = softmax(S/(|q||k|)*temp)
//   K5 E        = W_out @ blockdiag(attn)
//   K6 out      = E (192x192) @ v (192x16384)       [bf16 WMMA, dbl-buffered LDS]
// f32->bf16 staging uses native __bf16 conversions so the backend can emit
// packed v_cvt_*bf16_f32 instead of manual bit-twiddling VALU sequences.
// Workspace (floats):
//   [qkv_raw 75497472][qkv_dw 75497472][attn 36864][E 294912][Sg 65536][nqg 1536][nkg 1536]
// ---------------------------------------------------------------------------

typedef __bf16 v16bf  __attribute__((ext_vector_type(16)));
typedef __bf16 bf16x8 __attribute__((ext_vector_type(8)));
typedef float  f32x8  __attribute__((ext_vector_type(8)));

// Convert 8 f32 (two float4) to a bf16x8 with native RNE conversions.
__device__ __forceinline__ bf16x8 cvt8(const float4 lo, const float4 hi) {
    bf16x8 r;
    r[0] = (__bf16)lo.x; r[1] = (__bf16)lo.y;
    r[2] = (__bf16)lo.z; r[3] = (__bf16)lo.w;
    r[4] = (__bf16)hi.x; r[5] = (__bf16)hi.y;
    r[6] = (__bf16)hi.z; r[7] = (__bf16)hi.w;
    return r;
}

// Build a 16-element WMMA fragment from two 16B-aligned 8-element LDS runs.
__device__ __forceinline__ v16bf load_frag(const __bf16* p0, const __bf16* p1) {
    union U { bf16x8 h[2]; v16bf v; } u;
    u.h[0] = *reinterpret_cast<const bf16x8*>(p0);
    u.h[1] = *reinterpret_cast<const bf16x8*>(p1);
    return u.v;
}

// ---------------------------------------------------------------------------
// K1 / K6: C[b] (Mtot x N) = A[b] (Mtot x K) * B[b] (K x N), row-major f32.
// bf16 WMMA, f32 accumulate. Block tile 64(M) x 128(N), 8 waves = 4(M) x 2(N).
// Double-buffered LDS + register staging: one barrier per 32-wide K chunk,
// global float4 loads for chunk i+1 overlap the WMMAs of chunk i.
// ---------------------------------------------------------------------------
__global__ __launch_bounds__(256) void gemm_bf16_wmma(
    const float* __restrict__ A, const float* __restrict__ B,
    float* __restrict__ C, int Mtot, int N, int K,
    long a_bstride, long b_bstride)
{
    const int b      = blockIdx.z;
    const int m_base = blockIdx.y * 64;
    const int n_base = blockIdx.x * 128;
    A += (long)b * a_bstride;
    B += (long)b * b_bstride;
    C += (long)b * (long)Mtot * N;

    __shared__ __bf16 As[2][64][40];   // 32 used + pad (80B stride, 16B aligned)
    __shared__ __bf16 Bs[2][32][136];  // 128 used + pad (272B stride, 16B aligned)

    const int tid    = threadIdx.x;
    const int wave   = tid >> 5;
    const int lane   = tid & 31;
    const int m_off  = (wave & 3) * 16;
    const int n_offw = (wave >> 2) * 64;

    // staging assignment: A 64x32 -> 8 floats/thread; B 32x128 -> 16 floats/thread
    const int ra = tid >> 2, sa = (tid & 3) * 8;
    const int rb = tid >> 3, sb = (tid & 7) * 16;
    const float* Arow = A + (long)(m_base + ra) * K + sa;
    const float* Brow = B + (long)rb * N + n_base + sb;

    float4 a0, a1, b0, b1, b2, b3;
    auto gload = [&](int k0) {
        const float* ap = Arow + k0;
        a0 = *(const float4*)(ap);
        a1 = *(const float4*)(ap + 4);
        const float* bp = Brow + (long)k0 * N;
        b0 = *(const float4*)(bp);
        b1 = *(const float4*)(bp + 4);
        b2 = *(const float4*)(bp + 8);
        b3 = *(const float4*)(bp + 12);
    };
    auto lstore = [&](int buf) {
        *(bf16x8*)&As[buf][ra][sa]     = cvt8(a0, a1);
        *(bf16x8*)&Bs[buf][rb][sb]     = cvt8(b0, b1);
        *(bf16x8*)&Bs[buf][rb][sb + 8] = cvt8(b2, b3);
    };

    f32x8 acc[4];
#pragma unroll
    for (int t = 0; t < 4; ++t)
#pragma unroll
        for (int r = 0; r < 8; ++r) acc[t][r] = 0.f;

    gload(0);
    lstore(0);
    __syncthreads();

    const int nk = K >> 5;
    for (int i = 0; i < nk; ++i) {
        const int cur = i & 1;
        if (i + 1 < nk) gload((i + 1) << 5);

        // A frag: lanes 0-15 -> row M, K runs [0..7]+[16..23]; lanes 16-31 -> +8
        const int arow = m_off + (lane & 15);
        const int kh   = (lane >> 4) << 3;
        v16bf afrag = load_frag(&As[cur][arow][kh], &As[cur][arow][kh + 16]);
#pragma unroll
        for (int t = 0; t < 4; ++t) {
            const int n0 = n_offw + t * 16;   // B frag: lane = N column, 16 K rows in A's k order
            v16bf bfrag;
#pragma unroll
            for (int s = 0; s < 16; ++s)
                bfrag[s] = Bs[cur][kh + (s & 7) + ((s >> 3) << 4)][n0 + (lane & 15)];
            acc[t] = __builtin_amdgcn_wmma_f32_16x16x32_bf16(
                false, afrag, false, bfrag, (short)0, acc[t], false, false);
        }

        if (i + 1 < nk) lstore((i + 1) & 1);
        __syncthreads();
    }

    // C/D layout: lane<16 -> N=lane, reg r -> M=r ; lane>=16 -> N=lane-16, M=r+8
    const int n_l  = lane & 15;
    const int m_hi = (lane >> 4) << 3;
#pragma unroll
    for (int t = 0; t < 4; ++t) {
        const int n = n_base + n_offw + t * 16 + n_l;
#pragma unroll
        for (int r = 0; r < 8; ++r)
            C[(long)(m_base + m_off + m_hi + r) * N + n] = acc[t][r];
    }
}

// ---------------------------------------------------------------------------
// K2: depthwise 3x3, SAME zero padding. 4 consecutive pixels per thread.
// ---------------------------------------------------------------------------
__global__ __launch_bounds__(256) void dwconv3x3(
    const float* __restrict__ in, const float* __restrict__ wdw,
    float* __restrict__ out)
{
    const int idx = blockIdx.x * 256 + threadIdx.x;   // total 8*576*4096 groups
    const int q   = idx & 4095;
    const int bo  = idx >> 12;                        // b*576 + ch
    const int ch  = bo % 576;
    const int x4  = (q & 31) * 4;
    const int y   = q >> 5;

    const float* src = in + (long)bo * 16384;
    const float* w   = wdw + ch * 9;

    float s0 = 0.f, s1 = 0.f, s2 = 0.f, s3 = 0.f;
#pragma unroll
    for (int dy = -1; dy <= 1; ++dy) {
        const int yy = y + dy;
        if (yy < 0 || yy > 127) continue;
        const float* row = src + yy * 128;
        const float4 c  = *(const float4*)(row + x4);
        const float left  = (x4 > 0)   ? row[x4 - 1] : 0.f;
        const float right = (x4 < 124) ? row[x4 + 4] : 0.f;
        const float w0 = w[(dy + 1) * 3 + 0];
        const float w1 = w[(dy + 1) * 3 + 1];
        const float w2 = w[(dy + 1) * 3 + 2];
        s0 += w0 * left + w1 * c.x + w2 * c.y;
        s1 += w0 * c.x  + w1 * c.y + w2 * c.z;
        s2 += w0 * c.y  + w1 * c.z + w2 * c.w;
        s3 += w0 * c.z  + w1 * c.w + w2 * right;
    }
    float4 o; o.x = s0; o.y = s1; o.z = s2; o.w = s3;
    *(float4*)(out + (long)bo * 16384 + y * 128 + x4) = o;
}

// ---------------------------------------------------------------------------
// K3: partial channel attention. Grid (head, batch, ksplit). Each block does a
// 4096-wide slice of the K=16384 reduction: S_partial = q k^T via bf16 WMMA
// (24 rows padded to 32 -> 2x2 16-tiles, K split 32/wave per 256-chunk) and
// f32 |q|^2 / |k|^2 partials, accumulated into global buffers with atomics.
// ---------------------------------------------------------------------------
#define ACHUNK 256
#define KSPLIT 4
__global__ __launch_bounds__(256) void attn_partial(
    const float* __restrict__ qkv, float* __restrict__ Sg,
    float* __restrict__ nqg, float* __restrict__ nkg)
{
    const int h  = blockIdx.x;
    const int b  = blockIdx.y;
    const int ks = blockIdx.z;
    const long HW    = 16384;
    const long kspan = HW / KSPLIT;          // 4096
    const long kbeg  = (long)ks * kspan;
    const float* qbase = qkv + ((long)b * 576 + h * 24) * HW;
    const float* kbase = qkv + ((long)b * 576 + 192 + h * 24) * HW;

    __shared__ __bf16 Qs[32][ACHUNK + 8]; // rows 24..31 stay zero
    __shared__ __bf16 KsT[ACHUNK][40];    // cols 24..31 stay zero

    const int tid  = threadIdx.x;
    const int wave = tid >> 5;
    const int lane = tid & 31;

    // zero pad regions once (never rewritten inside the loop)
    for (int i = tid; i < 8 * (ACHUNK + 8); i += 256)
        Qs[24 + i / (ACHUNK + 8)][i % (ACHUNK + 8)] = (__bf16)0.f;
    for (int i = tid; i < ACHUNK * 16; i += 256)
        KsT[i / 16][24 + (i % 16)] = (__bf16)0.f;

    // loader mapping: 192 threads = 24 rows x 8 segments of 32 consecutive K
    const bool loader = (tid < 192);
    const int  lr     = tid >> 3;
    const int  loff   = (tid & 7) * 32;
    float qsum = 0.f, ksum = 0.f;

    f32x8 acc[4];
#pragma unroll
    for (int t = 0; t < 4; ++t)
#pragma unroll
        for (int r = 0; r < 8; ++r) acc[t][r] = 0.f;

    for (long k0 = kbeg; k0 < kbeg + kspan; k0 += ACHUNK) {
        __syncthreads();
        if (loader) {
            const float* qr = qbase + (long)lr * HW + k0 + loff;
            const float* kr = kbase + (long)lr * HW + k0 + loff;
#pragma unroll
            for (int g = 0; g < 4; ++g) {               // 8 floats per group
                const float4 q0 = *(const float4*)(qr + g * 8);
                const float4 q1 = *(const float4*)(qr + g * 8 + 4);
                const float4 kv0 = *(const float4*)(kr + g * 8);
                const float4 kv1 = *(const float4*)(kr + g * 8 + 4);
                qsum += q0.x*q0.x + q0.y*q0.y + q0.z*q0.z + q0.w*q0.w
                      + q1.x*q1.x + q1.y*q1.y + q1.z*q1.z + q1.w*q1.w;
                ksum += kv0.x*kv0.x + kv0.y*kv0.y + kv0.z*kv0.z + kv0.w*kv0.w
                      + kv1.x*kv1.x + kv1.y*kv1.y + kv1.z*kv1.z + kv1.w*kv1.w;
                *(bf16x8*)&Qs[lr][loff + g * 8] = cvt8(q0, q1);  // row-major q
                KsT[loff + g * 8 + 0][lr] = (__bf16)kv0.x;       // transposed k
                KsT[loff + g * 8 + 1][lr] = (__bf16)kv0.y;
                KsT[loff + g * 8 + 2][lr] = (__bf16)kv0.z;
                KsT[loff + g * 8 + 3][lr] = (__bf16)kv0.w;
                KsT[loff + g * 8 + 4][lr] = (__bf16)kv1.x;
                KsT[loff + g * 8 + 5][lr] = (__bf16)kv1.y;
                KsT[loff + g * 8 + 6][lr] = (__bf16)kv1.z;
                KsT[loff + g * 8 + 7][lr] = (__bf16)kv1.w;
            }
        }
        __syncthreads();

        const int sub = wave * 32;
        const int ar  = lane & 15;
        const int kh  = sub + ((lane >> 4) << 3);
#pragma unroll
        for (int mi = 0; mi < 2; ++mi) {
            v16bf af = load_frag(&Qs[mi * 16 + ar][kh], &Qs[mi * 16 + ar][kh + 16]);
#pragma unroll
            for (int ni = 0; ni < 2; ++ni) {
                v16bf bf;
#pragma unroll
                for (int s = 0; s < 16; ++s)
                    bf[s] = KsT[kh + (s & 7) + ((s >> 3) << 4)][ni * 16 + ar];
                acc[mi * 2 + ni] = __builtin_amdgcn_wmma_f32_16x16x32_bf16(
                    false, af, false, bf, (short)0, acc[mi * 2 + ni], false, false);
            }
        }
    }

    // accumulate partials into global buffers
    float* Sb = Sg + ((long)b * 8 + h) * 1024;        // [32][32]
    const int n_l  = lane & 15;
    const int m_hi = (lane >> 4) << 3;
#pragma unroll
    for (int mi = 0; mi < 2; ++mi)
#pragma unroll
        for (int ni = 0; ni < 2; ++ni)
#pragma unroll
            for (int r = 0; r < 8; ++r)
                atomicAdd(&Sb[(mi * 16 + m_hi + r) * 32 + ni * 16 + n_l],
                          acc[mi * 2 + ni][r]);
    if (loader) {
        atomicAdd(&nqg[(b * 8 + h) * 24 + lr], qsum);
        atomicAdd(&nkg[(b * 8 + h) * 24 + lr], ksum);
    }
}

// ---------------------------------------------------------------------------
// K4: normalize + temperature + softmax over 24 cols; one thread per row.
// ---------------------------------------------------------------------------
__global__ __launch_bounds__(32) void attn_final(
    const float* __restrict__ Sg, const float* __restrict__ nqg,
    const float* __restrict__ nkg, const float* __restrict__ temperature,
    float* __restrict__ attn_out)
{
    const int h = blockIdx.x, b = blockIdx.y;
    const int c = threadIdx.x;
    if (c >= 24) return;
    const float tmp = temperature[h];
    const float* S  = Sg + ((long)b * 8 + h) * 1024 + c * 32;
    const float* nq = nqg + (b * 8 + h) * 24;
    const float* nk = nkg + (b * 8 + h) * 24;
    const float qn  = fmaxf(sqrtf(nq[c]), 1e-12f);
    float row[24];
    float mx = -1e30f;
    for (int d = 0; d < 24; ++d) {
        const float kn = fmaxf(sqrtf(nk[d]), 1e-12f);
        const float v  = S[d] / (qn * kn) * tmp;
        row[d] = v;
        mx = fmaxf(mx, v);
    }
    float sum = 0.f;
    for (int d = 0; d < 24; ++d) { row[d] = __expf(row[d] - mx); sum += row[d]; }
    const float inv = 1.f / sum;
    float* o = attn_out + (((long)b * 8 + h) * 24 + c) * 24;
    for (int d = 0; d < 24; ++d) o[d] = row[d] * inv;
}

// ---------------------------------------------------------------------------
// K5: E[b] = W_out @ blockdiag(attn[b]); E[o,h*24+d] = sum_c W[o,h*24+c]*attn[h,c,d]
// ---------------------------------------------------------------------------
__global__ __launch_bounds__(256) void build_E(
    const float* __restrict__ wout, const float* __restrict__ attn,
    float* __restrict__ E)
{
    const int b   = blockIdx.y;
    const int idx = blockIdx.x * 256 + threadIdx.x;   // 192*192 = 36864
    const int o   = idx / 192;
    const int col = idx % 192;
    const int h   = col / 24, d = col % 24;

    const float* a    = attn + (((long)b * 8 + h) * 24) * 24 + d;
    const float* wrow = wout + o * 192 + h * 24;
    float s = 0.f;
#pragma unroll
    for (int c = 0; c < 24; ++c) s += wrow[c] * a[c * 24];
    E[(long)b * 36864 + idx] = s;
}

// ---------------------------------------------------------------------------
extern "C" void kernel_launch(void* const* d_in, const int* in_sizes, int n_in,
                              void* d_out, int out_size, void* d_ws, size_t ws_size,
                              hipStream_t stream) {
    const float* x     = (const float*)d_in[0];  // (8,192,128,128)
    const float* w_qkv = (const float*)d_in[1];  // (576,192)
    const float* w_dw  = (const float*)d_in[2];  // (576,1,3,3)
    const float* w_out = (const float*)d_in[3];  // (192,192)
    const float* temp  = (const float*)d_in[4];  // (8,)
    float*       out   = (float*)d_out;          // (8,192,128,128)

    const long HW = 16384, Cc = 192, C3 = 576, Bb = 8;
    float* ws       = (float*)d_ws;              // needs ~606 MB
    float* qkv_raw  = ws;
    float* qkv_dw   = qkv_raw + Bb * C3 * HW;
    float* attn     = qkv_dw  + Bb * C3 * HW;
    float* E        = attn + Bb * 8 * 24 * 24;           // +36864
    float* Sg       = E + Bb * Cc * Cc;                  // +294912
    float* nqg      = Sg + Bb * 8 * 32 * 32;             // +65536
    float* nkg      = nqg + Bb * 8 * 24;                 // +1536

    const dim3 blk(256);

    // K0: zero the partial-S / norm accumulators (capturable async memset)
    hipMemsetAsync(Sg, 0, (Bb * 8 * 32 * 32 + 2 * Bb * 8 * 24) * sizeof(float),
                   stream);

    // K1: qkv = W_qkv @ x
    gemm_bf16_wmma<<<dim3(128, 9, 8), blk, 0, stream>>>(
        w_qkv, x, qkv_raw, 576, (int)HW, 192, 0L, Cc * HW);

    // K2: depthwise 3x3 (8*576*4096 / 256 = 73728 blocks)
    dwconv3x3<<<dim3(73728), blk, 0, stream>>>(qkv_raw, w_dw, qkv_dw);

    // K3: partial attention logits + norms, K split 4 ways for occupancy
    attn_partial<<<dim3(8, 8, KSPLIT), blk, 0, stream>>>(qkv_dw, Sg, nqg, nkg);

    // K4: softmax
    attn_final<<<dim3(8, 8), dim3(32), 0, stream>>>(Sg, nqg, nkg, temp, attn);

    // K5: fold projection through attention
    build_E<<<dim3(144, 8), blk, 0, stream>>>(w_out, attn, E);

    // K6: out = E @ v   (v = qkv_dw channels 384..575, batch stride 576*HW)
    gemm_bf16_wmma<<<dim3(128, 3, 8), blk, 0, stream>>>(
        E, qkv_dw + 2 * Cc * HW, out, 192, (int)HW, 192,
        (long)(192 * 192), C3 * HW);
}
